// BidLSTM_52080773431931
// MI455X (gfx1250) — matrix operand order repaired and checked by reading, not compiled, not run on hardware
//
#include <hip/hip_runtime.h>
#include <hip/hip_bf16.h>
#include <cstdint>
#include <cstddef>

// ---------------------------------------------------------------------------
// Bidirectional LSTM on MI455X (gfx1250), wave32 + V_WMMA_F32_16X16X32_BF16.
// Per-timestep fused GEMM z = [x_t, h_prev] @ [Wx; Wh]  (K = D + H = 1536).
//  - A panel (32 batch rows x 1536 bf16) staged to LDS once per step with
//    GLOBAL_LOAD_ASYNC_TO_LDS_B128 (ASYNCcnt) -> 4-way reuse across gate waves.
//  - B weights pre-transposed bf16 [col][k]; streamed from global (L2-resident)
//    with a 1-stage register pipeline so loads overlap WMMAs.
// ---------------------------------------------------------------------------

typedef __bf16 bf16_t;
typedef __attribute__((ext_vector_type(16))) __bf16 v16bf;
typedef __attribute__((ext_vector_type(8)))  float  v8f;

constexpr int B = 64, T = 512, D = 512, H = 1024;
constexpr int G4H = 4 * H;       // 4096 stacked gate columns (i,f,o,g)
constexpr int NT  = 2;           // 16-wide n-tiles per wave (per gate)
constexpr int AK  = D + H;       // unified K = 1536
constexpr int ASTR = AK + 8;     // A-panel LDS row stride (+16B => 4-bank skew)

// dynamic LDS layout
constexpr size_t ALDS_BYTES = (size_t)32 * ASTR * sizeof(bf16_t);   // 98,816 B
constexpr size_t ZLDS_BYTES = (size_t)4 * 32 * (16 * NT) * sizeof(float); // 16 KB
constexpr size_t SMEM_BYTES = ALDS_BYTES + ZLDS_BYTES;              // ~115 KB

// ---- workspace layout ----
constexpr size_t XBF_ELEMS  = (size_t)B * T * D;            // bf16 inputs
constexpr size_t WXT_ELEMS  = (size_t)2 * G4H * D;          // bf16 WxT [dir][4H][D]
constexpr size_t WHT_ELEMS  = (size_t)2 * G4H * H;          // bf16 WhT [dir][4H][H]
constexpr size_t HBUF_ELEMS = (size_t)2 * 2 * B * H;        // bf16 h [dir][par][B][H]
constexpr size_t CBUF_ELEMS = (size_t)2 * B * H;            // f32  c [dir][B][H]

constexpr size_t XBF_BYTES  = XBF_ELEMS  * sizeof(bf16_t);
constexpr size_t WXT_BYTES  = WXT_ELEMS  * sizeof(bf16_t);
constexpr size_t WHT_BYTES  = WHT_ELEMS  * sizeof(bf16_t);
constexpr size_t HBUF_BYTES = HBUF_ELEMS * sizeof(bf16_t);

// ---------------- fragment loaders (wave32 layouts, ISA 7.12.2) -------------

// A 16x32 bf16 fragment out of the LDS A panel (row-major, stride ASTR).
// lanes 0-15: k = k0 + 0..15; lanes 16-31: k = k0 + 16..31.
__device__ __forceinline__ v16bf load_a_lds(const bf16_t* __restrict__ row,
                                            int k0, int lh) {
  v16bf a;
  ((uint4*)&a)[0] = *(const uint4*)(row + k0 + lh * 16);       // ds_load_b128
  ((uint4*)&a)[1] = *(const uint4*)(row + k0 + lh * 16 + 8);   // ds_load_b128
  return a;
}

// B 32x16 bf16 fragment from transposed weights WT[col][k] (row-major, ld=K).
// lanes 0-15 hold k = k0+0..15, lanes 16-31 hold k = k0+16..31 (contiguous 32B).
__device__ __forceinline__ v16bf load_b_frag(const bf16_t* __restrict__ colp,
                                             int k0, int lh) {
  v16bf b;
  const uint4* p = (const uint4*)(colp + k0 + lh * 16);
  ((uint4*)&b)[0] = p[0];
  ((uint4*)&b)[1] = p[1];
  return b;
}

// one 16B-per-lane async global->LDS copy (gfx1250, tracked by ASYNCcnt)
__device__ __forceinline__ void async_g2l_b128(uint32_t lds_off,
                                               const bf16_t* g) {
  asm volatile("global_load_async_to_lds_b128 %0, %1, off"
               :: "v"(lds_off), "v"(g) : "memory");
}

__device__ __forceinline__ float fast_sigmoid(float x) {
  return 1.0f / (1.0f + __expf(-x));
}
__device__ __forceinline__ float fast_tanh(float x) {
  return 1.0f - 2.0f / (__expf(2.0f * x) + 1.0f);   // stable: -> +-1
}

// ---------------------------- prep kernels ---------------------------------

__global__ void k_f32_to_bf16(const float* __restrict__ src,
                              bf16_t* __restrict__ dst, size_t n) {
  size_t i = (size_t)blockIdx.x * blockDim.x + threadIdx.x;
  if (i < n) dst[i] = (bf16_t)src[i];
}

// dst[g][k] = (bf16) src[k][g]   (src row-major [K, G]; coalesced reads)
__global__ void k_transpose_bf16(const float* __restrict__ src,
                                 bf16_t* __restrict__ dst, int K, int G) {
  size_t i = (size_t)blockIdx.x * blockDim.x + threadIdx.x;
  if (i >= (size_t)K * G) return;
  int k = (int)(i / G);
  int g = (int)(i % G);
  dst[(size_t)g * K + k] = (bf16_t)src[i];
}

__global__ void k_zero_state(bf16_t* __restrict__ h, float* __restrict__ c) {
  size_t i = (size_t)blockIdx.x * blockDim.x + threadIdx.x;
  if (i < HBUF_ELEMS) h[i] = (bf16_t)0.0f;
  if (i < CBUF_ELEMS) c[i] = 0.0f;
}

// ---------------------------- LSTM step kernel ------------------------------
// grid = (H/(16*NT), B/32, 2 dirs), block = 256 (8 waves).
// wave w: gate = w&3, m-block = w>>2.  Each wave: 16(M) x 16*NT(N), K=1536.

__global__ __launch_bounds__(256)
void k_lstm_step(const bf16_t* __restrict__ xbf,   // [B][T][D] bf16
                 const bf16_t* __restrict__ wxt,   // [2][4H][D] bf16 (transposed)
                 const bf16_t* __restrict__ wht,   // [2][4H][H] bf16 (transposed)
                 const float*  __restrict__ bias_f,
                 const float*  __restrict__ bias_b,
                 bf16_t*       __restrict__ hbuf,  // [2][2][B][H] bf16 ping-pong
                 float*        __restrict__ cbuf,  // [2][B][H] f32
                 float*        __restrict__ out,   // [B][T][2H] f32
                 int t) {
  extern __shared__ char smem[];
  bf16_t* alds = (bf16_t*)smem;                       // [32][ASTR] A panel
  float*  zlds = (float*)(smem + ALDS_BYTES);         // [4][32][16*NT] gate z
#define ZL(g, m, n) zlds[(((g) * 32 + (m)) * (16 * NT)) + (n)]

  const int tid  = threadIdx.x;
  const int lane = tid & 31;
  const int wave = tid >> 5;
  const int gate = wave & 3;                       // 0=i 1=f 2=o 3=g
  const int mblk = wave >> 2;                      // 0..1
  const int lh   = lane >> 4;
  const int ln   = lane & 15;

  const int nblk = blockIdx.x;                     // hidden-col block (16*NT wide)
  const int mgrp = blockIdx.y;                     // batch block of 32
  const int dir  = blockIdx.z;
  const int tt   = dir ? (T - 1 - t) : t;
  const int par  = t & 1;

  const bf16_t* wxtd = wxt + (size_t)dir * G4H * D;
  const bf16_t* whtd = wht + (size_t)dir * G4H * H;
  const float*  bias = dir ? bias_b : bias_f;
  const bf16_t* hsrc = hbuf + (size_t)(dir * 2 + par)       * B * H;
  bf16_t*       hdst = hbuf + (size_t)(dir * 2 + (par ^ 1)) * B * H;

  // ---- stage A panel [32 rows x 1536] = x_t || h_prev into LDS (async) -----
  {
    const uint32_t abase = (uint32_t)(uintptr_t)alds;     // LDS byte offset
    constexpr int CPR = AK / 8;                           // 16B chunks per row
    for (int c = tid; c < 32 * CPR; c += 256) {           // 24 iters, uniform
      const int row = c / CPR;
      const int k   = (c % CPR) * 8;
      const int bb  = mgrp * 32 + row;
      const bf16_t* g = (k < D) ? (xbf + ((size_t)bb * T + tt) * D + k)
                                : (hsrc + (size_t)bb * H + (k - D));
      async_g2l_b128(abase + (uint32_t)(row * ASTR + k) * 2u, g);
    }
  }

  // ---- B pointers (computed while async copies are in flight) --------------
  const bf16_t* wxcol[NT];
  const bf16_t* whcol[NT];
#pragma unroll
  for (int nt = 0; nt < NT; ++nt) {
    const int col = gate * H + nblk * (16 * NT) + nt * 16 + ln;
    wxcol[nt] = wxtd + (size_t)col * D;
    whcol[nt] = whtd + (size_t)col * H;
  }
  const bf16_t* arow = alds + (size_t)(mblk * 16 + ln) * ASTR;

  v8f acc[NT] = {};

  // prime the B register pipeline (global loads overlap the async staging)
  v16bf bcur[NT];
#pragma unroll
  for (int nt = 0; nt < NT; ++nt) bcur[nt] = load_b_frag(wxcol[nt], 0, lh);

  // A panel must be resident before ds reads
  asm volatile("s_wait_asynccnt 0x0" ::: "memory");
  __syncthreads();

  // ---- x contribution (K = 0..D); last iteration preloads first h block ----
#pragma unroll 4
  for (int k = 0; k < D; k += 32) {
    const int kn = k + 32;
    v16bf bn[NT];
    if (kn < D) {
      __builtin_prefetch(wxcol[0] + kn + 32, 0, 3);
#pragma unroll
      for (int nt = 0; nt < NT; ++nt) bn[nt] = load_b_frag(wxcol[nt], kn, lh);
    } else {
#pragma unroll
      for (int nt = 0; nt < NT; ++nt) bn[nt] = load_b_frag(whcol[nt], 0, lh);
    }
    v16bf a = load_a_lds(arow, k, lh);
#pragma unroll
    for (int nt = 0; nt < NT; ++nt)
      acc[nt] = __builtin_amdgcn_wmma_f32_16x16x32_bf16(
          false, a, false, bcur[nt], (short)0, acc[nt], false, false);
#pragma unroll
    for (int nt = 0; nt < NT; ++nt) bcur[nt] = bn[nt];
  }

  // ---- h contribution (K = 0..H); next-index clamped, body stays uniform ---
#pragma unroll 4
  for (int k = 0; k < H; k += 32) {
    const int kn = (k + 32 < H) ? (k + 32) : 0;      // last preload discarded
    __builtin_prefetch(whcol[0] + kn + 32, 0, 3);
    v16bf bn[NT];
#pragma unroll
    for (int nt = 0; nt < NT; ++nt) bn[nt] = load_b_frag(whcol[nt], kn, lh);
    v16bf a = load_a_lds(arow, D + k, lh);
#pragma unroll
    for (int nt = 0; nt < NT; ++nt)
      acc[nt] = __builtin_amdgcn_wmma_f32_16x16x32_bf16(
          false, a, false, bcur[nt], (short)0, acc[nt], false, false);
#pragma unroll
    for (int nt = 0; nt < NT; ++nt) bcur[nt] = bn[nt];
  }

  // scatter z (+bias) into LDS; D-matrix layout: m = r + 8*lh, n = lane&15
#pragma unroll
  for (int nt = 0; nt < NT; ++nt) {
    const float bv = bias[gate * H + nblk * (16 * NT) + nt * 16 + ln];
#pragma unroll
    for (int r = 0; r < 8; ++r)
      ZL(gate, mblk * 16 + r + 8 * lh, nt * 16 + ln) = acc[nt][r] + bv;
  }
  __syncthreads();

  // ---- gates + state update: 32 x (16*NT) h-values per workgroup ----
  for (int e = tid; e < 32 * 16 * NT; e += 256) {
    const int m   = e / (16 * NT);
    const int n   = e % (16 * NT);
    const int b   = mgrp * 32 + m;
    const int col = nblk * (16 * NT) + n;

    const float iv = fast_sigmoid(ZL(0, m, n));
    const float fv = fast_sigmoid(ZL(1, m, n));
    const float ov = fast_sigmoid(ZL(2, m, n));
    const float gv = fast_tanh(ZL(3, m, n));

    const size_t cidx = ((size_t)dir * B + b) * H + col;
    const float cn = fv * cbuf[cidx] + iv * gv;
    cbuf[cidx] = cn;
    const float hn = ov * fast_tanh(cn);

    hdst[(size_t)b * H + col] = (bf16_t)hn;                       // next-step A
    out[((size_t)b * T + tt) * (2 * H) + (size_t)dir * H + col] = hn;
  }
#undef ZL
}

// ------------------------------- launcher -----------------------------------

extern "C" void kernel_launch(void* const* d_in, const int* in_sizes, int n_in,
                              void* d_out, int out_size, void* d_ws, size_t ws_size,
                              hipStream_t stream) {
  (void)in_sizes; (void)n_in; (void)out_size; (void)ws_size;

  const float* x    = (const float*)d_in[0];
  const float* Wx_f = (const float*)d_in[1];
  const float* Wh_f = (const float*)d_in[2];
  const float* b_fw = (const float*)d_in[3];
  const float* Wx_b = (const float*)d_in[4];
  const float* Wh_b = (const float*)d_in[5];
  const float* b_bw = (const float*)d_in[6];
  float* out = (float*)d_out;

  char*   ws  = (char*)d_ws;
  bf16_t* xbf = (bf16_t*)ws;
  bf16_t* wxt = (bf16_t*)(ws + XBF_BYTES);
  bf16_t* wht = (bf16_t*)(ws + XBF_BYTES + WXT_BYTES);
  bf16_t* hb  = (bf16_t*)(ws + XBF_BYTES + WXT_BYTES + WHT_BYTES);
  float*  cb  = (float*) (ws + XBF_BYTES + WXT_BYTES + WHT_BYTES + HBUF_BYTES);

  // one-time (per call) prep: bf16 inputs, transposed bf16 weights, zero state
  k_f32_to_bf16<<<(unsigned)((XBF_ELEMS + 255) / 256), 256, 0, stream>>>(x, xbf, XBF_ELEMS);

  const size_t nwx = (size_t)D * G4H, nwh = (size_t)H * G4H;
  k_transpose_bf16<<<(unsigned)((nwx + 255) / 256), 256, 0, stream>>>(Wx_f, wxt, D, G4H);
  k_transpose_bf16<<<(unsigned)((nwx + 255) / 256), 256, 0, stream>>>(Wx_b, wxt + nwx, D, G4H);
  k_transpose_bf16<<<(unsigned)((nwh + 255) / 256), 256, 0, stream>>>(Wh_f, wht, H, G4H);
  k_transpose_bf16<<<(unsigned)((nwh + 255) / 256), 256, 0, stream>>>(Wh_b, wht + nwh, H, G4H);
  k_zero_state<<<(unsigned)((HBUF_ELEMS + 255) / 256), 256, 0, stream>>>(hb, cb);

  // sequential scan: launch boundary = device-wide sync between timesteps
  dim3 grid(H / (16 * NT), B / 32, 2), blk(256);
  for (int t = 0; t < T; ++t)
    k_lstm_step<<<grid, blk, SMEM_BYTES, stream>>>(xbf, wxt, wht, b_fw, b_bw,
                                                   hb, cb, out, t);
}
